// PointNetMSGRandomSample_42236708389459
// MI455X (gfx1250) — compile-verified, matrix operand order checked by reading, not run on hardware
//
#include <hip/hip_runtime.h>

#define BB 4
#define NN 8192
#define SS 2048

typedef __attribute__((ext_vector_type(16))) _Float16 v16h;
typedef __attribute__((ext_vector_type(8)))  _Float16 v8h;
typedef __attribute__((ext_vector_type(8)))  float    v8f;

// ---------------------------------------------------------------- new_xyz gather
__global__ void newxyz_kernel(const float* __restrict__ xyz, const int* __restrict__ sidx,
                              float* __restrict__ out) {
  int i = blockIdx.x * blockDim.x + threadIdx.x;   // over B*S
  if (i >= BB * SS) return;
  int b = i / SS;
  int gi = sidx[i];
  const float* p = xyz + ((size_t)b * NN + gi) * 3;
  out[3 * i + 0] = p[0];
  out[3 * i + 1] = p[1];
  out[3 * i + 2] = p[2];
}

// ---------------------------------------------------------------- f32 -> padded f16 weights
__global__ void prep_weights_kernel(const float* __restrict__ w, _Float16* __restrict__ wh,
                                    int c_out, int c_in, int Kp) {
  int i = blockIdx.x * blockDim.x + threadIdx.x;
  if (i >= c_out * Kp) return;
  int r = i / Kp, c = i % Kp;
  wh[i] = (c < c_in) ? (_Float16)w[r * c_in + c] : (_Float16)0.0f;
}

__global__ void zero_stats_kernel(float* __restrict__ p, int n) {
  int i = blockIdx.x * blockDim.x + threadIdx.x;
  if (i < n) p[i] = 0.0f;
}

// ---------------------------------------------------------------- ball query: one WAVE per query
// 32 waves/block share one LDS copy of xyz[b] (96KB of the 320KB WGP LDS).
// Each iteration a wave tests 32 consecutive points; ballot+popcount compacts
// in-radius lanes IN INDEX ORDER (matches reference first-k-by-index semantics).
// Count updates are wave-uniform -> early exit keeps EXEC coherent.
__global__ void __launch_bounds__(1024) ballquery_kernel(const float* __restrict__ xyz,
                                                         const float* __restrict__ nxyz,
                                                         int* __restrict__ g0,
                                                         int* __restrict__ g1) {
  extern __shared__ float sx[];                 // NN*3 floats = 96KB
  int b = blockIdx.y;
  const float* xb = xyz + (size_t)b * NN * 3;
  for (int i = threadIdx.x; i < NN * 3; i += blockDim.x) sx[i] = xb[i];
  __syncthreads();

  int wave = threadIdx.x >> 5;                  // 0..31
  int lane = threadIdx.x & 31;
  int s  = blockIdx.x * 32 + wave;              // query index, always < SS
  int bs = b * SS + s;
  float cx = nxyz[3 * bs + 0], cy = nxyz[3 * bs + 1], cz = nxyz[3 * bs + 2];

  int* o0 = g0 + (size_t)bs * 16;
  int* o1 = g1 + (size_t)bs * 32;
  int c0 = 0, c1 = 0;                           // wave-uniform saturating counts
  const float r0sq = 0.1f * 0.1f;
  const float r1sq = 0.2f * 0.2f;
  unsigned lt = (1u << lane) - 1u;

  for (int nb = 0; nb < NN; nb += 32) {
    int n = nb + lane;
    float dx = sx[3 * n + 0] - cx;
    float dy = sx[3 * n + 1] - cy;
    float dz = sx[3 * n + 2] - cz;
    float d = dx * dx + dy * dy + dz * dz;
    bool in1 = (d <= r1sq);
    bool in0 = (d <= r0sq);
    unsigned m1 = __builtin_amdgcn_ballot_w32(in1);
    unsigned m0 = __builtin_amdgcn_ballot_w32(in0);
    if (in1) {
      int pos = c1 + __popc(m1 & lt);
      if (pos < 32) o1[pos] = n;
    }
    if (in0) {
      int pos = c0 + __popc(m0 & lt);
      if (pos < 16) o0[pos] = n;
    }
    c1 = min(32, c1 + __popc(m1));
    c0 = min(16, c0 + __popc(m0));
    if (c1 == 32 && c0 == 16) break;
  }
  // pad with first in-radius index (query point itself guarantees >=1 hit)
  int f0 = o0[0];
  int f1 = o1[0];
  for (int t = c0 + lane; t < 16; t += 32) o0[t] = f0;
  for (int t = c1 + lane; t < 32; t += 32) o1[t] = f1;
}

// ---------------------------------------------------------------- grouped input (6ch -> padded 32, f16)
__global__ void build_input_kernel(const float* __restrict__ xyz, const float* __restrict__ feat,
                                   const float* __restrict__ nxyz, const int* __restrict__ gidx,
                                   _Float16* __restrict__ X, int K, int M) {
  int i = blockIdx.x * blockDim.x + threadIdx.x;   // over B*S*K
  if (i >= M) return;
  int bs = i / K;
  int b  = bs / SS;
  int gi = gidx[i];
  const float* p = xyz  + ((size_t)b * NN + gi) * 3;
  const float* f = feat + ((size_t)b * NN + gi) * 3;
  const float* c = nxyz + (size_t)bs * 3;
  v8h lo;
  lo[0] = (_Float16)f[0];
  lo[1] = (_Float16)f[1];
  lo[2] = (_Float16)f[2];
  lo[3] = (_Float16)(p[0] - c[0]);
  lo[4] = (_Float16)(p[1] - c[1]);
  lo[5] = (_Float16)(p[2] - c[2]);
  lo[6] = (_Float16)0.0f;
  lo[7] = (_Float16)0.0f;
  v8h z = {};
  v8h* row = (v8h*)(X + (size_t)i * 32);     // 64B-aligned
  row[0] = lo;
  row[1] = z;
  row[2] = z;
  row[3] = z;
}

// ---------------------------------------------------------------- WMMA GEMM: Y = X * W^T + b (f16 in, f32 acc)
// One wave -> 16x16 tile. A frag: lane<16 row M=lane, K(p)= (p<8?p:p+8)+8*half.
// W stored (c_out x Kp) row-major so the B (K x N) fragment loads contiguously.
// Consecutive waves in a block share an m-tile -> A rows hit L0/L2.
__global__ void __launch_bounds__(256) gemm_wmma_kernel(
    const _Float16* __restrict__ X, const _Float16* __restrict__ Wh,
    const float* __restrict__ bias, _Float16* __restrict__ Y,
    int Kp, int C, int totalWaves) {
  int wave = blockIdx.x * 8 + (threadIdx.x >> 5);
  if (wave >= totalWaves) return;                 // wave-uniform: EXEC all-1s inside
  int lane = threadIdx.x & 31;
  int ntiles = C >> 4;
  int m0 = (wave / ntiles) << 4;
  int n0 = (wave % ntiles) << 4;
  int half = lane >> 4;
  int lr   = lane & 15;

  const _Float16* ap = X  + (size_t)(m0 + lr) * Kp + half * 8;
  const _Float16* bp = Wh + (size_t)(n0 + lr) * Kp + half * 8;

  v8f acc = {};
  for (int kk = 0; kk < Kp; kk += 32) {
    v8h alo = *(const v8h*)(ap + kk);
    v8h ahi = *(const v8h*)(ap + kk + 16);
    v8h blo = *(const v8h*)(bp + kk);
    v8h bhi = *(const v8h*)(bp + kk + 16);
    v16h a, bm;
#pragma unroll
    for (int i = 0; i < 8; ++i) {
      a[i] = alo[i];  a[i + 8] = ahi[i];
      bm[i] = blo[i]; bm[i + 8] = bhi[i];
    }
    acc = __builtin_amdgcn_wmma_f32_16x16x32_f16(false, a, false, bm,
                                                 (short)0, acc, false, false);
  }

  int n = n0 + lr;
  float bv = bias[n];
#pragma unroll
  for (int r = 0; r < 8; ++r) {
    int m = m0 + r + half * 8;
    Y[(size_t)m * C + n] = (_Float16)(acc[r] + bv);
  }
}

// ---------------------------------------------------------------- per-channel sum/sumsq (LDS-reduced)
// Fixed 256 blocks x 256 threads; thread t owns column (t % C), strides rows.
// Coalesced f16 reads (Y is L2-resident), ds_add_f32 block reduction,
// then only 2*C global atomics per block.
__global__ void __launch_bounds__(256) reduce_stats_kernel(const _Float16* __restrict__ Y,
                                                           float* __restrict__ slab,
                                                           int M, int C, int R) {
  __shared__ float ls[512];
  int t = threadIdx.x;
  for (int i = t; i < 2 * C; i += 256) ls[i] = 0.0f;
  __syncthreads();
  int gt = blockIdx.x * 256 + t;
  int c  = gt % C;
  int r0 = gt / C;
  float s = 0.0f, s2 = 0.0f;
  for (int r = r0; r < M; r += R) {
    float v = (float)Y[(size_t)r * C + c];
    s += v;
    s2 += v * v;
  }
  atomicAdd(&ls[c], s);
  atomicAdd(&ls[C + c], s2);
  __syncthreads();
  for (int i = t; i < C; i += 256) {
    atomicAdd(&slab[i], ls[i]);
    atomicAdd(&slab[256 + i], ls[C + i]);
  }
}

// ---------------------------------------------------------------- BN stats -> effective scale/shift
__global__ void finalize_stats_kernel(float* __restrict__ slab, const float* __restrict__ gamma,
                                      const float* __restrict__ beta, float invM, int C) {
  int c = blockIdx.x * blockDim.x + threadIdx.x;
  if (c >= C) return;
  float mean = slab[c] * invM;
  float var  = slab[256 + c] * invM - mean * mean;
  float sc   = gamma[c] * rsqrtf(var + 1e-5f);
  slab[512 + c] = sc;
  slab[768 + c] = beta[c] - mean * sc;
}

// ---------------------------------------------------------------- BN + PReLU in place, 8 halves/thread
__global__ void bn_prelu_kernel(_Float16* __restrict__ Y, const float* __restrict__ slab,
                                const float* __restrict__ a, size_t total8, int C) {
  size_t i = (size_t)blockIdx.x * blockDim.x + threadIdx.x;
  if (i >= total8) return;
  v8h v = ((const v8h*)Y)[i];
  int c0 = (int)((i * 8) % (size_t)C);          // C is a multiple of 8
  float al = a[0];
  v8h r;
#pragma unroll
  for (int e = 0; e < 8; ++e) {
    float x = (float)v[e];
    x = slab[512 + c0 + e] * x + slab[768 + c0 + e];
    r[e] = (_Float16)(x >= 0.0f ? x : al * x);
  }
  ((v8h*)Y)[i] = r;
}

// ---------------------------------------------------------------- BN + PReLU + max over group (into concat buffer)
__global__ void bn_prelu_max_kernel(const _Float16* __restrict__ Y, const float* __restrict__ slab,
                                    const float* __restrict__ a, _Float16* __restrict__ pooled,
                                    int BSn, int K, int C, int pStride, int offc) {
  int i = blockIdx.x * blockDim.x + threadIdx.x;   // over BS*C
  if (i >= BSn * C) return;
  int bs = i / C, c = i % C;
  float sc = slab[512 + c], sh = slab[768 + c], al = a[0];
  float mx = -3.4e38f;
  const _Float16* p = Y + (size_t)bs * K * C + c;
  for (int j = 0; j < K; ++j) {
    float v = sc * (float)p[(size_t)j * C] + sh;
    v = (v >= 0.0f) ? v : al * v;
    mx = fmaxf(mx, v);
  }
  pooled[(size_t)bs * pStride + offc + c] = (_Float16)mx;
}

// ---------------------------------------------------------------- final BN + PReLU + transpose to (B,256,S) f32
__global__ void bn_prelu_out_kernel(const _Float16* __restrict__ Y, const float* __restrict__ slab,
                                    const float* __restrict__ a, float* __restrict__ out) {
  int i = blockIdx.x * blockDim.x + threadIdx.x;   // over B*S*256
  if (i >= BB * SS * 256) return;
  int c = i % 256;
  int bs = i / 256;
  int b = bs / SS, s = bs % SS;
  float v = slab[512 + c] * (float)Y[i] + slab[768 + c];
  v = (v >= 0.0f) ? v : a[0] * v;
  out[((size_t)b * 256 + c) * SS + s] = v;
}

// ================================================================ host orchestration
extern "C" void kernel_launch(void* const* d_in, const int* in_sizes, int n_in,
                              void* d_out, int out_size, void* d_ws, size_t ws_size,
                              hipStream_t stream) {
  (void)in_sizes; (void)n_in; (void)out_size; (void)ws_size;
  const float* xyz  = (const float*)d_in[0];
  const float* feat = (const float*)d_in[1];
  const int*   sidx = (const int*)d_in[37];   // params flatten: 7 layers x (w,b,gamma,beta,a)
  auto Wp = [&](int l) { return (const float*)d_in[2 + 5 * l + 0]; };
  auto Bp = [&](int l) { return (const float*)d_in[2 + 5 * l + 1]; };
  auto Gp = [&](int l) { return (const float*)d_in[2 + 5 * l + 2]; };
  auto Ep = [&](int l) { return (const float*)d_in[2 + 5 * l + 3]; };
  auto Ap = [&](int l) { return (const float*)d_in[2 + 5 * l + 4]; };

  static const int cin_[7]  = {6, 32, 32,   6, 64, 64,   192};
  static const int kp_[7]   = {32, 32, 32,  32, 64, 64,  192};
  static const int cout_[7] = {32, 32, 64,  64, 64, 128, 256};
  static const int M_[7]    = {131072, 131072, 131072, 262144, 262144, 262144, 8192};

  char* ws = (char*)d_ws;
  size_t off = 0;
  auto alloc = [&](size_t bytes) -> char* {
    char* p = ws + off;
    off = (off + bytes + 255) & ~(size_t)255;
    return p;
  };
  int*      gidx0  = (int*)alloc((size_t)BB * SS * 16 * 4);
  int*      gidx1  = (int*)alloc((size_t)BB * SS * 32 * 4);
  _Float16* bufA   = (_Float16*)alloc((size_t)262144 * 128 * 2);
  _Float16* bufB   = (_Float16*)alloc((size_t)262144 * 128 * 2);
  _Float16* pooled = (_Float16*)alloc((size_t)BB * SS * 192 * 2);
  float*    stats  = (float*)alloc((size_t)7 * 1024 * 4);
  _Float16* wh[7];
  for (int l = 0; l < 7; ++l) wh[l] = (_Float16*)alloc((size_t)cout_[l] * kp_[l] * 2);

  float* nxyz = (float*)d_out;                       // (B,S,3)
  float* outF = (float*)d_out + (size_t)BB * SS * 3; // (B,256,S)

  // phase 0: gathers + weight conversion + stats reset
  newxyz_kernel<<<(BB * SS + 255) / 256, 256, 0, stream>>>(xyz, sidx, nxyz);
  for (int l = 0; l < 7; ++l)
    prep_weights_kernel<<<(cout_[l] * kp_[l] + 255) / 256, 256, 0, stream>>>(
        Wp(l), wh[l], cout_[l], cin_[l], kp_[l]);
  zero_stats_kernel<<<(7 * 1024 + 255) / 256, 256, 0, stream>>>(stats, 7 * 1024);

  // phase 1: wave-per-query ball query, both radii in one LDS-staged scan
  ballquery_kernel<<<dim3(SS / 32, BB), 1024, (size_t)NN * 3 * sizeof(float), stream>>>(
      xyz, nxyz, gidx0, gidx1);

  // phase 2: per-scale grouped MLPs (WMMA) + BN/PReLU + max pool
  for (int sc = 0; sc < 2; ++sc) {
    int K = (sc == 0) ? 16 : 32;
    const int* gidx = (sc == 0) ? gidx0 : gidx1;
    int l0 = sc * 3;
    int M = M_[l0];
    build_input_kernel<<<(M + 255) / 256, 256, 0, stream>>>(xyz, feat, nxyz, gidx, bufA, K, M);
    _Float16* cur = bufA;
    _Float16* nxt = bufB;
    for (int j = 0; j < 3; ++j) {
      int l = l0 + j;
      float* slab = stats + (size_t)l * 1024;
      int waves = (M / 16) * (cout_[l] / 16);
      gemm_wmma_kernel<<<(waves + 7) / 8, 256, 0, stream>>>(
          cur, wh[l], Bp(l), nxt, kp_[l], cout_[l], waves);
      reduce_stats_kernel<<<256, 256, 0, stream>>>(nxt, slab, M, cout_[l], 65536 / cout_[l]);
      finalize_stats_kernel<<<(cout_[l] + 255) / 256, 256, 0, stream>>>(
          slab, Gp(l), Ep(l), 1.0f / (float)M, cout_[l]);
      if (j < 2) {
        size_t tot8 = ((size_t)M * cout_[l]) / 8;
        bn_prelu_kernel<<<(int)((tot8 + 255) / 256), 256, 0, stream>>>(
            nxt, slab, Ap(l), tot8, cout_[l]);
        _Float16* t = cur; cur = nxt; nxt = t;
      } else {
        int offc = (sc == 0) ? 0 : 64;
        bn_prelu_max_kernel<<<(BB * SS * cout_[l] + 255) / 256, 256, 0, stream>>>(
            nxt, slab, Ap(l), pooled, BB * SS, K, cout_[l], 192, offc);
      }
    }
  }

  // phase 3: final 192->256 layer + BN + PReLU + transpose
  {
    int l = 6;
    float* slab = stats + (size_t)l * 1024;
    int M = M_[l];
    int waves = (M / 16) * (cout_[l] / 16);
    gemm_wmma_kernel<<<(waves + 7) / 8, 256, 0, stream>>>(
        pooled, wh[l], Bp(l), bufA, kp_[l], cout_[l], waves);
    reduce_stats_kernel<<<256, 256, 0, stream>>>(bufA, slab, M, cout_[l], 65536 / cout_[l]);
    finalize_stats_kernel<<<1, 256, 0, stream>>>(slab, Gp(l), Ep(l), 1.0f / (float)M, cout_[l]);
    bn_prelu_out_kernel<<<(BB * SS * 256 + 255) / 256, 256, 0, stream>>>(
        bufA, slab, Ap(l), outF);
  }
}
